// InterBranch_40003325395144
// MI455X (gfx1250) — matrix operand order repaired and checked by reading, not compiled, not run on hardware
//
#include <hip/hip_runtime.h>
#include <hip/hip_bf16.h>

typedef __attribute__((ext_vector_type(16))) _Float16 v16h;
typedef __attribute__((ext_vector_type(8)))  float    v8f;

#define D_DIM 128
#define NHEAD 8
#define HDIM  16

// ---------------------------------------------------------------------------
// Kernel 1: initialize mutable feature buffer f[(N+1) x 128]; row N = zeros.
// ---------------------------------------------------------------------------
__global__ void init_f_kernel(float4* __restrict__ f4,
                              const float4* __restrict__ feats4,
                              long n4, long tot4) {
  long i = (long)blockIdx.x * blockDim.x + threadIdx.x;
  if (i >= tot4) return;
  float4 z; z.x = z.y = z.z = z.w = 0.0f;
  f4[i] = (i < n4) ? feats4[i] : z;
}

// ---------------------------------------------------------------------------
// Kernel 2: one topological level: f[child] += f[parent]  (128 floats / node)
// ---------------------------------------------------------------------------
__global__ void level_kernel(float* __restrict__ f,
                             const int* __restrict__ child,
                             const int* __restrict__ parent,
                             int M, int N) {
  long t = (long)blockIdx.x * blockDim.x + threadIdx.x;
  int node = (int)(t >> 5);      // 32 float4 per node row
  int c    = (int)(t & 31);
  if (node >= M) return;
  int ci = child[node];
  if (ci >= N) return;           // dummy padding entry
  int pi = parent[node];         // may be N (root) -> zero row
  const float4 a = ((const float4*)(f + (long)pi * D_DIM))[c];
  float4* dst = (float4*)(f + (long)ci * D_DIM) + c;
  float4 v = *dst;
  v.x += a.x; v.y += a.y; v.z += a.z; v.w += a.w;
  *dst = v;
}

// ---------------------------------------------------------------------------
// Kernel 3: gather x = f[branch_idx] and QKV projection via WMMA f16->f32.
// One wave (32 lanes) computes a 16x16 output tile; K=128 in 4 chunks of 32.
// Outputs: q (scaled by 1/sqrt(hd)) and k as [B,H,S,16] f16; v transposed as
// [B,H,16,S] f16 so the attention B-operand columns are contiguous.
// ---------------------------------------------------------------------------
__global__ __launch_bounds__(32)
void qkv_kernel(const float* __restrict__ f,       // (N+1) x 128
                const float* __restrict__ W,       // 384 x 128 (in_proj_weight)
                const float* __restrict__ bias,    // 384
                const int*   __restrict__ bidx,    // B*S (dummy = N)
                _Float16* __restrict__ qb,
                _Float16* __restrict__ kb,
                _Float16* __restrict__ vT,
                int S) {
  const int tile     = blockIdx.x;
  const int row_tile = tile / 24;          // over B*S/16 row tiles
  const int col_tile = tile % 24;          // over 384/16 output cols
  const int lane = threadIdx.x;
  const int m    = lane & 15;
  const int half = lane >> 4;

  const int g   = row_tile * 16 + m;       // global row (b*S + s) for A operand
  const int idx = bidx[g];                 // f row (N => zeros)
  const float* xrow = f + (long)idx * D_DIM;

  const int n     = lane & 15;
  const int n_out = col_tile * 16 + n;     // W row for B operand

  v8f acc = {};
#pragma unroll
  for (int kc = 0; kc < 4; ++kc) {
    // A operand: 16x32 f16. element e: k = kc*32 + half*16 + e
    v16h a;
#pragma unroll
    for (int e = 0; e < 8; ++e) {
      a[e]     = (_Float16)xrow[kc * 32 + half * 16 + e];
      a[e + 8] = (_Float16)xrow[kc * 32 + half * 16 + 8 + e];
    }
    // B operand: 32x16 f16. lane(n, khi=half): element e -> W[n_out][kc*32 + half*16 + e]
    const float* wrow = W + (long)n_out * D_DIM + kc * 32 + half * 16;
    v16h b;
#pragma unroll
    for (int e = 0; e < 16; ++e) b[e] = (_Float16)wrow[e];
    acc = __builtin_amdgcn_wmma_f32_16x16x32_f16(false, a, false, b,
                                                 (short)0, acc, false, false);
  }

  const float bv = bias[n_out];
#pragma unroll
  for (int r = 0; r < 8; ++r) {
    const int M  = r + 8 * half;           // C/D row layout
    const int gg = row_tile * 16 + M;
    const int b  = gg / S;
    const int s  = gg % S;
    const float val = acc[r] + bv;
    if (n_out < 128) {                     // Q (pre-scale by 1/sqrt(16))
      const int h = n_out / HDIM, dd = n_out % HDIM;
      qb[(((long)b * NHEAD + h) * S + s) * HDIM + dd] = (_Float16)(val * 0.25f);
    } else if (n_out < 256) {              // K
      const int h = (n_out - 128) / HDIM, dd = (n_out - 128) % HDIM;
      kb[(((long)b * NHEAD + h) * S + s) * HDIM + dd] = (_Float16)val;
    } else {                               // V (transposed)
      const int h = (n_out - 256) / HDIM, dd = (n_out - 256) % HDIM;
      vT[(((long)b * NHEAD + h) * HDIM + dd) * S + s] = (_Float16)val;
    }
  }
}

// ---------------------------------------------------------------------------
// Kernel 4: attention. One wave per (b, h, 16-query tile).
// scores (16 x S) in LDS f32, exact softmax (additive 1/0 mask), P@V via WMMA.
// ---------------------------------------------------------------------------
__global__ __launch_bounds__(32)
void attn_kernel(const _Float16* __restrict__ qb,
                 const _Float16* __restrict__ kb,
                 const _Float16* __restrict__ vT,
                 const int* __restrict__ lengths,
                 float* __restrict__ obuf,       // [B, S, 128] f32
                 int S) {
  const int qtiles = S / 16;
  const int bh = blockIdx.x / qtiles;
  const int qt = blockIdx.x % qtiles;
  const int b  = bh >> 3;
  const int h  = bh & 7;
  const int lane = threadIdx.x;
  const int m    = lane & 15;
  const int half = lane >> 4;
  const int len  = lengths[b];

  __shared__ float    sc[16][512];         // 32 KB score tile
  __shared__ _Float16 pr[16][512];         // 16 KB probability tile

  const _Float16* qbase = qb + (long)bh * S * HDIM;
  const _Float16* kbase = kb + (long)bh * S * HDIM;
  const _Float16* vbase = vT + (long)bh * HDIM * S;

  // A operand (Q tile, K padded 16->32 with zeros)
  v16h aq = {};
  {
    const _Float16* qrow = qbase + (long)(qt * 16 + m) * HDIM;
#pragma unroll
    for (int e = 0; e < 16; ++e) if (half == 0) aq[e] = qrow[e];
  }

  // ---- scores: 32 WMMAs (16 queries x 16 keys each, K padded to 32) ----
  for (int kt = 0; kt < qtiles; ++kt) {
    v16h bk = {};
    if (half == 0) {                       // lanes 0-15 hold k=0..15; 16-31 are k=16..31 (zero pad)
      const _Float16* krow = kbase + (long)(kt * 16 + m) * HDIM;
#pragma unroll
      for (int e = 0; e < 16; ++e) bk[e] = krow[e];
    }
    v8f c = {};
    c = __builtin_amdgcn_wmma_f32_16x16x32_f16(false, aq, false, bk,
                                               (short)0, c, false, false);
    const int n   = lane & 15;
    const int key = kt * 16 + n;
    const float kvalid = (key < len) ? 1.0f : 0.0f;
#pragma unroll
    for (int r = 0; r < 8; ++r) {
      const int M = r + 8 * half;
      const float qvalid = (qt * 16 + M) < len ? 1.0f : 0.0f;
      sc[M][key] = c[r] + qvalid * kvalid; // additive float mask, per reference
    }
  }
  __syncthreads();

  // ---- exact softmax over S keys: row = lane&15, column half = lane>>4 ----
  {
    const int row = lane & 15;
    const int c0  = half * (S / 2);
    float mx = -3.0e38f;
    for (int c = 0; c < S / 2; ++c) mx = fmaxf(mx, sc[row][c0 + c]);
    mx = fmaxf(mx, __shfl_xor(mx, 16, 32));
    float sum = 0.0f;
    for (int c = 0; c < S / 2; ++c) {
      const float e = __expf(sc[row][c0 + c] - mx);
      sum += e;
      pr[row][c0 + c] = (_Float16)e;
    }
    sum += __shfl_xor(sum, 16, 32);
    const float inv = 1.0f / sum;
    for (int c = 0; c < S / 2; ++c)
      pr[row][c0 + c] = (_Float16)((float)pr[row][c0 + c] * inv);
  }
  __syncthreads();

  // ---- O = P @ V : 16 WMMAs over key chunks of 32 ----
  v8f o = {};
  for (int kc = 0; kc < S / 32; ++kc) {
    v16h pa;
    const _Float16* p0 = &pr[m][kc * 32 + half * 16];
    const _Float16* p1 = &pr[m][kc * 32 + half * 16 + 8];
#pragma unroll
    for (int e = 0; e < 8; ++e) { pa[e] = p0[e]; pa[e + 8] = p1[e]; }
    v16h bv;
    const _Float16* vrow = vbase + (long)(lane & 15) * S + kc * 32 + half * 16;
#pragma unroll
    for (int e = 0; e < 16; ++e) bv[e] = vrow[e];
    o = __builtin_amdgcn_wmma_f32_16x16x32_f16(false, pa, false, bv,
                                               (short)0, o, false, false);
  }

  const int n = lane & 15;
#pragma unroll
  for (int r = 0; r < 8; ++r) {
    const int M = r + 8 * half;
    obuf[((long)b * S + qt * 16 + M) * D_DIM + h * HDIM + n] = o[r];
  }
}

// ---------------------------------------------------------------------------
// Kernel 5: osum[b][d] = sum_s obuf[b][s][d]
// ---------------------------------------------------------------------------
__global__ void reduce_kernel(const float* __restrict__ obuf,
                              float* __restrict__ osum, int B, int S) {
  const int t = blockIdx.x * blockDim.x + threadIdx.x;
  if (t >= B * D_DIM) return;
  const int b = t / D_DIM, d = t % D_DIM;
  float acc = 0.0f;
  const float* p = obuf + (long)b * S * D_DIM + d;
  for (int s = 0; s < S; ++s) acc += p[(long)s * D_DIM];
  osum[t] = acc;
}

// ---------------------------------------------------------------------------
// Kernel 6: out[b][j] = osum[b][:] . Wout[j][:] + S * bout[j]
// ---------------------------------------------------------------------------
__global__ void outproj_kernel(const float* __restrict__ osum,
                               const float* __restrict__ Wout,
                               const float* __restrict__ bout,
                               float* __restrict__ out, int S) {
  const int b = blockIdx.x;
  const int j = threadIdx.x;
  const float* x = osum + (long)b * D_DIM;
  const float* w = Wout + (long)j * D_DIM;
  float acc = 0.0f;
#pragma unroll 4
  for (int d = 0; d < D_DIM; ++d) acc += x[d] * w[d];
  out[b * D_DIM + j] = acc + (float)S * bout[j];
}

// ---------------------------------------------------------------------------
static inline size_t align256(size_t x) { return (x + 255) & ~(size_t)255; }

extern "C" void kernel_launch(void* const* d_in, const int* in_sizes, int n_in,
                              void* d_out, int out_size, void* d_ws, size_t ws_size,
                              hipStream_t stream) {
  const float* feats = (const float*)d_in[0];
  const float* Win   = (const float*)d_in[1];
  const float* bin   = (const float*)d_in[2];
  const float* Wout  = (const float*)d_in[3];
  const float* bout  = (const float*)d_in[4];
  const int*   child = (const int*)d_in[5];
  const int*   parent= (const int*)d_in[6];
  const int*   bidx  = (const int*)d_in[7];
  const int*   lens  = (const int*)d_in[8];

  const long N = (long)in_sizes[0] / D_DIM;
  int L = 0; while (((1L << L) - 1) < N) ++L;          // topological levels
  const int M  = in_sizes[5] / L;                      // padded level width
  const int B  = in_sizes[8];
  const int S  = in_sizes[7] / B;
  const int BS = B * S;

  // workspace carve-up
  char* ws = (char*)d_ws;
  size_t off = 0;
  float* f = (float*)(ws + off);            off = align256(off + (size_t)(N + 1) * D_DIM * sizeof(float));
  _Float16* qb = (_Float16*)(ws + off);     off = align256(off + (size_t)BS * D_DIM * sizeof(_Float16));
  _Float16* kb = (_Float16*)(ws + off);     off = align256(off + (size_t)BS * D_DIM * sizeof(_Float16));
  _Float16* vT = (_Float16*)(ws + off);     off = align256(off + (size_t)BS * D_DIM * sizeof(_Float16));
  float* obuf = (float*)(ws + off);         off = align256(off + (size_t)BS * D_DIM * sizeof(float));
  float* osum = (float*)(ws + off);         off = align256(off + (size_t)B * D_DIM * sizeof(float));
  (void)ws_size; (void)n_in; (void)out_size;

  // 1) init f (copy feats, zero dummy row N)
  {
    const long tot4 = (N + 1) * (D_DIM / 4);
    const long n4   = N * (D_DIM / 4);
    const int blocks = (int)((tot4 + 255) / 256);
    init_f_kernel<<<blocks, 256, 0, stream>>>((float4*)f, (const float4*)feats, n4, tot4);
  }

  // 2) tree prefix-add, level by level
  {
    const long work = (long)M * 32;
    const int blocks = (int)((work + 255) / 256);
    for (int l = 0; l < L; ++l) {
      level_kernel<<<blocks, 256, 0, stream>>>(f, child + (long)l * M,
                                               parent + (long)l * M, M, (int)N);
    }
  }

  // 3) gather + QKV projection (WMMA)
  qkv_kernel<<<(BS / 16) * 24, 32, 0, stream>>>(f, Win, bin, bidx, qb, kb, vT, S);

  // 4) attention (WMMA scores + WMMA P@V)
  attn_kernel<<<B * NHEAD * (S / 16), 32, 0, stream>>>(qb, kb, vT, lens, obuf, S);

  // 5) sum over sequence
  reduce_kernel<<<(B * D_DIM + 255) / 256, 256, 0, stream>>>(obuf, osum, B, S);

  // 6) output projection + bias*S
  outproj_kernel<<<B, D_DIM, 0, stream>>>(osum, Wout, bout, (float*)d_out, S);
}
